// TypeSpecificGCNLayer_36885179138297
// MI455X (gfx1250) — matrix operand order repaired and checked by reading, not compiled, not run on hardware
//
#include <hip/hip_runtime.h>

// TypeSpecificGCNLayer on MI455X (gfx1250).
// out = bias + sum_r A_r @ (H @ W_r^T), bf16-pair-split WMMA with f32 accum.
//
// Roofline: streaming the three A matrices once = 1.2 GB -> 51 us at 23.3 TB/s.
// 3-term bf16 split (a_hi*b_hi + a_hi*b_lo + a_lo*b_hi) gives ~fp32 accuracy at
// 115 GF of bf16 WMMA, which the matrix pipes absorb within the memory floor.

#define NN      10000     // nodes
#define NPAD    10016     // NN padded up to multiple of 32 (K padding for WMMA)
#define DD      64        // in/out feature dim
#define NCHUNK  313       // ceil(NN / 32) K-chunks
#define NSLICE  16        // split-K slices
#define CPS     20        // ceil(NCHUNK / NSLICE) chunks per slice
#define NTILE   625       // NN / 16 row tiles
#define PFDIST  4         // prefetch distance in K-chunks (512 B per row)

typedef __attribute__((ext_vector_type(16))) __bf16          v16bf;
typedef __attribute__((ext_vector_type(8)))  float           v8f;
typedef __attribute__((ext_vector_type(8)))  unsigned int    v8u;
typedef __attribute__((ext_vector_type(4)))  float           v4f;

// ---------------------------------------------------------------------------
// Kernel 1: G_r = H @ W_r^T, stored transposed + padded + split into bf16
// hi/lo planes:  Gt layout = [3 relations][2 planes][64 cols][NPAD rows]
// Rows NN..NPAD-1 are zero so the main kernel's B loads need no K guard.
// ---------------------------------------------------------------------------
__global__ __launch_bounds__(256) void gcn_prep(
    const float* __restrict__ H,  const float* __restrict__ W1,
    const float* __restrict__ W2, const float* __restrict__ W3,
    unsigned short* __restrict__ Gt)
{
  const int k = blockIdx.x * 256 + threadIdx.x;   // tensor row (K index)
  if (k >= NPAD) return;
  const int c = blockIdx.y;                       // output column 0..63
  const int r = blockIdx.z;                       // relation 0..2
  const float* W = (r == 0) ? W1 : (r == 1) ? W2 : W3;

  float acc = 0.f;
  if (k < NN) {
    const float* h = H + (size_t)k * DD;
    const float* w = W + (size_t)c * DD;
#pragma unroll
    for (int d = 0; d < DD; ++d) acc = fmaf(h[d], w[d], acc);
  }
  const __bf16 hi = (__bf16)acc;                  // native cvt, RNE
  const __bf16 lo = (__bf16)(acc - (float)hi);

  const size_t idx = ((size_t)(r * 2) * DD + c) * NPAD + (size_t)k;
  Gt[idx]                     = __builtin_bit_cast(unsigned short, hi);
  Gt[idx + (size_t)DD * NPAD] = __builtin_bit_cast(unsigned short, lo);
}

// ---------------------------------------------------------------------------
// Kernel 2: main GEMM.  One wave = one 16-row x 64-col tile over one K slice.
// Per 32-wide K chunk: 3 relations x 4 col-tiles x 3 split terms = 36 WMMAs.
// ---------------------------------------------------------------------------
template <bool GUARD>
__device__ __forceinline__ void gcn_chunk(
    int u0, int half, int lane_lo,
    const float* __restrict__ r0, const float* __restrict__ r1,
    const float* __restrict__ r2,
    const unsigned short* __restrict__ Gt, v8f acc[4])
{
  const float* rows[3] = {r0, r1, r2};
  const int s0 = u0 + half * 16;    // A fragment: first K segment start
  const int kB = u0 + half * 16;    // B fragment: K base for this lane half

#pragma unroll
  for (int r = 0; r < 3; ++r) {
    // ---- A fragment: 16 f32 per lane (contiguous order, k = 16*half + s) ----
    float f[16];
#pragma unroll
    for (int seg = 0; seg < 2; ++seg) {
#pragma unroll
      for (int q = 0; q < 2; ++q) {
        const int cc = s0 + seg * 8 + q * 4;
        v4f v = {0.f, 0.f, 0.f, 0.f};
        if (!GUARD || cc < NN)      // NN % 4 == 0 -> whole float4 in/out
          v = __builtin_nontemporal_load(
                reinterpret_cast<const v4f*>(rows[r] + cc));
        f[seg * 8 + q * 4 + 0] = v[0];
        f[seg * 8 + q * 4 + 1] = v[1];
        f[seg * 8 + q * 4 + 2] = v[2];
        f[seg * 8 + q * 4 + 3] = v[3];
      }
    }
    // ---- split f32 -> bf16 hi + bf16 lo (native packed converts) ----
    v16bf ah, al;
#pragma unroll
    for (int i = 0; i < 16; ++i) {
      const __bf16 h = (__bf16)f[i];
      ah[i] = h;
      al[i] = (__bf16)(f[i] - (float)h);
    }

    const unsigned short* gh = Gt + (size_t)(r * 2 + 0) * DD * NPAD;
    const unsigned short* gl = Gt + (size_t)(r * 2 + 1) * DD * NPAD;

#pragma unroll
    for (int c4 = 0; c4 < 4; ++c4) {
      const int col = c4 * 16 + lane_lo;   // B col = lane % 16
      const v16bf bh = __builtin_bit_cast(v16bf,
          *reinterpret_cast<const v8u*>(gh + (size_t)col * NPAD + kB));
      const v16bf bl = __builtin_bit_cast(v16bf,
          *reinterpret_cast<const v8u*>(gl + (size_t)col * NPAD + kB));
      // 3-term product: ah*bh + ah*bl + al*bh  (~fp32 accuracy)
      acc[c4] = __builtin_amdgcn_wmma_f32_16x16x32_bf16(
          false, ah, false, bh, (short)0, acc[c4], false, false);
      acc[c4] = __builtin_amdgcn_wmma_f32_16x16x32_bf16(
          false, ah, false, bl, (short)0, acc[c4], false, false);
      acc[c4] = __builtin_amdgcn_wmma_f32_16x16x32_bf16(
          false, al, false, bh, (short)0, acc[c4], false, false);
    }
  }
}

__global__ __launch_bounds__(256) void gcn_main(
    const float* __restrict__ A1, const float* __restrict__ A2,
    const float* __restrict__ A3, const unsigned short* __restrict__ Gt,
    float* __restrict__ Part)
{
  const int wave    = threadIdx.x >> 5;
  const int lane    = threadIdx.x & 31;
  const int wid     = blockIdx.x * 8 + wave;   // 0 .. NTILE*NSLICE-1 (exact)
  const int tile    = wid / NSLICE;
  const int slice   = wid % NSLICE;
  const int m0      = tile * 16;
  const int lane_lo = lane & 15;
  const int half    = lane >> 4;

  const size_t rowoff = (size_t)(m0 + lane_lo) * NN;  // A row for this lane
  const float* r0 = A1 + rowoff;
  const float* r1 = A2 + rowoff;
  const float* r2 = A3 + rowoff;

  v8f acc[4] = {};

  const int cbeg = slice * CPS;
  int cend = cbeg + CPS;
  if (cend > NCHUNK) cend = NCHUNK;
  const int cfast = (cend > NCHUNK - 1) ? (NCHUNK - 1) : cend;

  for (int ci = cbeg; ci < cfast; ++ci) {
    // software prefetch: each lane pulls its row's chunk PFDIST ahead
    // (gfx1250 global_prefetch_b8; scalar bound check keeps it in-range)
    const int un = (ci + PFDIST) * 32;
    if (un + 32 <= NN) {
      __builtin_prefetch(r0 + un, 0, 3);
      __builtin_prefetch(r1 + un, 0, 3);
      __builtin_prefetch(r2 + un, 0, 3);
    }
    gcn_chunk<false>(ci * 32, half, lane_lo, r0, r1, r2, Gt, acc);
  }
  if (cend == NCHUNK)  // ragged final chunk (K 9984..10015)
    gcn_chunk<true>((NCHUNK - 1) * 32, half, lane_lo, r0, r1, r2, Gt, acc);

  // store f32 partials: Part[slice][row][col]  (C/D layout: M = 8*half + i)
  float* P = Part + (size_t)slice * NN * DD;
#pragma unroll
  for (int c4 = 0; c4 < 4; ++c4) {
    const int col = c4 * 16 + lane_lo;
#pragma unroll
    for (int i = 0; i < 8; ++i) {
      const int row = m0 + half * 8 + i;
      P[(size_t)row * DD + col] = acc[c4][i];
    }
  }
}

// ---------------------------------------------------------------------------
// Kernel 3: deterministic split-K reduction + bias.
// ---------------------------------------------------------------------------
__global__ __launch_bounds__(256) void gcn_reduce(
    const float* __restrict__ Part, const float* __restrict__ bias,
    float* __restrict__ out)
{
  const int idx = blockIdx.x * 256 + threadIdx.x;   // 0 .. NN*DD-1 (exact)
  float s = bias[idx & (DD - 1)];
#pragma unroll
  for (int p = 0; p < NSLICE; ++p)
    s += Part[(size_t)p * NN * DD + idx];
  out[idx] = s;
}

// ---------------------------------------------------------------------------
extern "C" void kernel_launch(void* const* d_in, const int* in_sizes, int n_in,
                              void* d_out, int out_size, void* d_ws, size_t ws_size,
                              hipStream_t stream)
{
  const float* H    = (const float*)d_in[0];
  const float* A1   = (const float*)d_in[1];
  const float* A2   = (const float*)d_in[2];
  const float* A3   = (const float*)d_in[3];
  const float* W1   = (const float*)d_in[4];
  const float* W2   = (const float*)d_in[5];
  const float* W3   = (const float*)d_in[6];
  const float* bias = (const float*)d_in[7];
  float* out = (float*)d_out;

  // workspace: [Gt bf16 hi/lo 7.7 MB][split-K partials 41 MB]
  unsigned short* Gt = (unsigned short*)d_ws;
  const size_t gt_bytes = (size_t)3 * 2 * DD * NPAD * sizeof(unsigned short);
  float* Part = (float*)((char*)d_ws + gt_bytes);

  gcn_prep<<<dim3((NPAD + 255) / 256, DD, 3), 256, 0, stream>>>(H, W1, W2, W3, Gt);
  gcn_main<<<dim3(NTILE * NSLICE / 8), 256, 0, stream>>>(A1, A2, A3, Gt, Part);
  gcn_reduce<<<dim3(NN * DD / 256), 256, 0, stream>>>(Part, bias, out);
}
